// MyModel_63823214019177
// MI455X (gfx1250) — matrix operand order repaired and checked by reading, not compiled, not run on hardware
//
#include <hip/hip_runtime.h>
#include <cstdint>
#include <cstddef>

// Problem dims
#define T_STEPS 512
#define BATCH   512
#define FDIM    578
#define HDIM    100
#define H4      400
#define VDIM    578
#define KPADX   608      // F padded to 19*32
#define KPADH   128      // H padded to 4*32
#define NT4H    25       // 400/16 N'-tiles

typedef __attribute__((ext_vector_type(16))) __bf16 v16bf;
typedef __attribute__((ext_vector_type(8)))  __bf16 v8bf;
typedef __attribute__((ext_vector_type(8)))  float  v8f;

union V16U { v16bf v; v8bf h[2]; };

// A fragment (16x32 bf16): lane<16 -> row m, K-halves {0..7, 16..23}; lane>=16 -> {8..15, 24..31}
__device__ __forceinline__ v16bf ldA(const __bf16* base, int rowStride, int lane, int kOff) {
  int m  = lane & 15;
  int kh = lane >> 4;
  const __bf16* p = base + m * rowStride + kOff + kh * 8;
  V16U t;
  t.h[0] = *(const v8bf*)p;          // K = kOff + kh*8 + 0..7
  t.h[1] = *(const v8bf*)(p + 16);   // K = kOff + 16 + kh*8 + 0..7
  return t.v;
}
// B fragment (32x16 bf16) from [N][K] (transposed) storage: lane<16 -> col n, K {0..7, 16..23}; lane>=16 -> {8..15, 24..31}
__device__ __forceinline__ v16bf ldB(const __bf16* base, int colStride, int n0, int lane, int kOff) {
  int n   = n0 + (lane & 15);
  int klo = (lane >> 4) * 8;
  const __bf16* p = base + n * colStride + kOff + klo;
  V16U t;
  t.h[0] = *(const v8bf*)p;
  t.h[1] = *(const v8bf*)(p + 16);
  return t.v;
}
__device__ __forceinline__ v8f wmma_bf16(v16bf a, v16bf b, v8f c) {
  return __builtin_amdgcn_wmma_f32_16x16x32_bf16(false, a, false, b, (short)0, c, false, false);
}
__device__ __forceinline__ float bperm_f32(float v, int srcLane) {
  return __int_as_float(__builtin_amdgcn_ds_bpermute(srcLane << 2, __float_as_int(v)));
}
__device__ __forceinline__ float sigf(float x) { return 1.0f / (1.0f + __expf(-x)); }

// --------------------------------------------------------------------------
// Kernel 0: W1 -> bf16, gate-interleaved column permutation (col' = h*4+g),
// transposed to [N'=400][K=608] with zero K-padding.
__global__ void __launch_bounds__(256) k_prep_w1(const float* __restrict__ W1,
                                                 __bf16* __restrict__ W1bf) {
  int idx = blockIdx.x * 256 + threadIdx.x;
  if (idx >= H4 * KPADX) return;
  int n = idx / KPADX, k = idx - n * KPADX;
  float v = (k < FDIM) ? W1[k * H4 + ((n & 3) * HDIM + (n >> 2))] : 0.0f;
  W1bf[idx] = (__bf16)v;
}

// --------------------------------------------------------------------------
// Kernel 1: XW[t*B+b][n'] = x[t,b,:] @ W1 + b1  (permuted columns, f32 out)
// M = T*B = 262144 (64 rows / workgroup), N = 400, K = 608 (chunks of 32).
__global__ void __launch_bounds__(256) k_gemm_xw(const float* __restrict__ X,
                                                 const __bf16* __restrict__ W1bf,
                                                 const float* __restrict__ b1,
                                                 float* __restrict__ XW) {
  __shared__ __align__(16) __bf16 Abuf[64 * 32];   // 4 KB
  __shared__ __align__(16) __bf16 Bbuf[H4 * 32];   // 25.6 KB
  const int tid  = threadIdx.x;
  const int lane = tid & 31;
  const int wv   = __builtin_amdgcn_readfirstlane(tid >> 5);
  const int mt   = wv >> 1;                 // 4 M-tiles of 16 rows
  const int half = wv & 1;                  // split 25 N-tiles: 13 + 12
  const int ntBase = half ? 13 : 0;
  const int ntCnt  = half ? 12 : 13;
  const size_t g0  = (size_t)blockIdx.x * 64;

  v8f acc[13];
  v8f zero = {0.f, 0.f, 0.f, 0.f, 0.f, 0.f, 0.f, 0.f};
#pragma unroll
  for (int j = 0; j < 13; ++j) acc[j] = zero;

  for (int kk = 0; kk < KPADX / 32; ++kk) {
    // stage A chunk [64][32] with f32->bf16 convert + K zero pad
    for (int e = tid; e < 64 * 32; e += 256) {
      int row = e >> 5, kl = e & 31;
      int k = kk * 32 + kl;
      float v = (k < FDIM) ? X[(g0 + (size_t)row) * FDIM + k] : 0.0f;
      Abuf[e] = (__bf16)v;
    }
    // stage B chunk [400][32] (bf16 copy; W1 bf16 stays L2-resident)
    {
      const uint32_t* src = (const uint32_t*)W1bf;
      uint32_t* dst = (uint32_t*)Bbuf;
      for (int d = tid; d < H4 * 16; d += 256) {
        int n = d >> 4, q = d & 15;
        dst[d] = src[n * (KPADX / 2) + kk * 16 + q];
      }
    }
    __syncthreads();
    // A fragment reused across all N-tiles; B fragments double-buffered so
    // the ds_read latency of B[j+1] overlaps the WMMA on B[j].
    v16bf a = ldA(Abuf + mt * 16 * 32, 32, lane, 0);
    v16bf bcur = ldB(Bbuf, 32, ntBase * 16, lane, 0);
#pragma unroll
    for (int j = 0; j < 13; ++j) {
      if (j < ntCnt) {
        v16bf bnext = bcur;
        if (j + 1 < ntCnt) bnext = ldB(Bbuf, 32, (ntBase + j + 1) * 16, lane, 0);
        acc[j] = wmma_bf16(a, bcur, acc[j]);
        bcur = bnext;
      }
    }
    __syncthreads();
  }
  // store (+ permuted b1 bias). C layout: VGPR r -> M=r (lanes 0-15) / r+8 (16-31), N=lane&15
  const int nlo   = lane & 15;
  const int rbase = (lane >> 4) << 3;
#pragma unroll
  for (int j = 0; j < 13; ++j) {
    if (j < ntCnt) {
      int n = (ntBase + j) * 16 + nlo;
      float bias = b1[(n & 3) * HDIM + (n >> 2)];
#pragma unroll
      for (int r = 0; r < 8; ++r) {
        size_t row = g0 + (size_t)(mt * 16 + rbase + r);
        XW[row * H4 + n] = acc[j][r] + bias;
      }
    }
  }
}

// --------------------------------------------------------------------------
// Kernel 2: persistent 2-layer LSTM recurrence. One workgroup per 16 batch
// rows (no inter-WGP sync needed: recurrence is batch-independent).
// LDS: U1,W2,U2 transposed/permuted bf16 [400][128] (307.2 KB) + h1/h2 tiles.
__global__ void __launch_bounds__(256) k_lstm(const float* __restrict__ XW,
                                              const float* __restrict__ U1,
                                              const float* __restrict__ W2,
                                              const float* __restrict__ U2,
                                              const float* __restrict__ b2,
                                              float* __restrict__ h2out) {
  extern __shared__ __align__(16) char smem[];
  __bf16* Wlds  = (__bf16*)smem;                 // [3][400][128]
  __bf16* h1buf = Wlds + 3 * H4 * KPADH;         // [16][128] (K-padded, zeros >=100)
  __bf16* h2buf = h1buf + 16 * KPADH;

  const int tid  = threadIdx.x;
  const int lane = tid & 31;
  const int wv   = __builtin_amdgcn_readfirstlane(tid >> 5);
  const int m0   = blockIdx.x * 16;

  // convert weights into LDS: [w][n'][k], permuted cols, zero K-pad
  for (int idx = tid; idx < 3 * H4 * KPADH; idx += 256) {
    int w = idx / (H4 * KPADH);
    int rem = idx - w * (H4 * KPADH);
    int n = rem >> 7;
    int k = rem & 127;
    const float* src = (w == 0) ? U1 : (w == 1) ? W2 : U2;
    float v = (k < HDIM) ? src[k * H4 + ((n & 3) * HDIM + (n >> 2))] : 0.0f;
    Wlds[idx] = (__bf16)v;
  }
  for (int idx = tid; idx < 2 * 16 * KPADH; idx += 256) h1buf[idx] = (__bf16)0.0f;
  __syncthreads();

  const __bf16* U1l = Wlds;
  const __bf16* W2l = Wlds + H4 * KPADH;
  const __bf16* U2l = Wlds + 2 * H4 * KPADH;

  const int nlo   = lane & 15;
  const int rbase = (lane >> 4) << 3;
  const int s     = lane & 3;       // gate role: 0=i 1=f 2=g 3=o (n' = h*4+g)
  const int gBase = lane & ~3;

  int   nts[4]; bool valT[4]; float biasv[4];
  float c1v[4][8], c2v[4][8];
#pragma unroll
  for (int i = 0; i < 4; ++i) {
    nts[i]  = wv + 8 * i;
    valT[i] = nts[i] < NT4H;
    int n = nts[i] * 16 + nlo;
    biasv[i] = valT[i] ? b2[(n & 3) * HDIM + (n >> 2)] : 0.0f;
#pragma unroll
    for (int r = 0; r < 8; ++r) { c1v[i][r] = 0.0f; c2v[i][r] = 0.0f; }
  }

  v8f   acc[4];
  v16bf aF[4];
  for (int t = 0; t < T_STEPS; ++t) {
    const float* xw = XW + ((size_t)t * BATCH + m0) * H4;
    // ---- layer 1 GEMM: z1 = XW[t] + h1 @ U1
    // Hoist the 4 K-step A fragments (h1 rows) once; reuse across all tiles.
#pragma unroll
    for (int ks = 0; ks < 4; ++ks) aF[ks] = ldA(h1buf, KPADH, lane, ks * 32);
#pragma unroll
    for (int i = 0; i < 4; ++i) {
      if (valT[i]) {
        int n = nts[i] * 16 + nlo;
#pragma unroll
        for (int r = 0; r < 8; ++r) acc[i][r] = xw[(size_t)(rbase + r) * H4 + n];
#pragma unroll
        for (int ks = 0; ks < 4; ++ks) {
          v16bf b = ldB(U1l, KPADH, nts[i] * 16, lane, ks * 32);
          acc[i] = wmma_bf16(aF[ks], b, acc[i]);
        }
      }
    }
    __syncthreads();                       // all h1 reads done
    // ---- layer 1 pointwise (gate-interleaved tiles -> 4-lane shuffles)
#pragma unroll
    for (int i = 0; i < 4; ++i) {
      if (valT[i]) {
#pragma unroll
        for (int r = 0; r < 8; ++r) {
          float z  = acc[i][r];
          float av = (s == 2) ? tanhf(z) : sigf(z);
          float gi = bperm_f32(av, gBase + 0);
          float gf = bperm_f32(av, gBase + 1);
          float gg = bperm_f32(av, gBase + 2);
          float go = bperm_f32(av, gBase + 3);
          float cn = gf * c1v[i][r] + gi * gg;
          c1v[i][r] = cn;
          float hn = go * tanhf(cn);
          if (s == 0) {
            int m = rbase + r;
            int h = (nts[i] << 2) + (nlo >> 2);
            h1buf[m * KPADH + h] = (__bf16)hn;
          }
        }
      }
    }
    __syncthreads();                       // new h1 visible
    // ---- layer 2 GEMM: z2 = b2 + h1 @ W2 + h2 @ U2
#pragma unroll
    for (int ks = 0; ks < 4; ++ks) aF[ks] = ldA(h1buf, KPADH, lane, ks * 32);
#pragma unroll
    for (int i = 0; i < 4; ++i) {
      if (valT[i]) {
#pragma unroll
        for (int r = 0; r < 8; ++r) acc[i][r] = biasv[i];
#pragma unroll
        for (int ks = 0; ks < 4; ++ks) {
          v16bf b = ldB(W2l, KPADH, nts[i] * 16, lane, ks * 32);
          acc[i] = wmma_bf16(aF[ks], b, acc[i]);
        }
      }
    }
#pragma unroll
    for (int ks = 0; ks < 4; ++ks) aF[ks] = ldA(h2buf, KPADH, lane, ks * 32);
#pragma unroll
    for (int i = 0; i < 4; ++i) {
      if (valT[i]) {
#pragma unroll
        for (int ks = 0; ks < 4; ++ks) {
          v16bf b = ldB(U2l, KPADH, nts[i] * 16, lane, ks * 32);
          acc[i] = wmma_bf16(aF[ks], b, acc[i]);
        }
      }
    }
    __syncthreads();                       // all h1/h2 reads done
    // ---- layer 2 pointwise (+ final-step f32 h2 output)
#pragma unroll
    for (int i = 0; i < 4; ++i) {
      if (valT[i]) {
#pragma unroll
        for (int r = 0; r < 8; ++r) {
          float z  = acc[i][r];
          float av = (s == 2) ? tanhf(z) : sigf(z);
          float gi = bperm_f32(av, gBase + 0);
          float gf = bperm_f32(av, gBase + 1);
          float gg = bperm_f32(av, gBase + 2);
          float go = bperm_f32(av, gBase + 3);
          float cn = gf * c2v[i][r] + gi * gg;
          c2v[i][r] = cn;
          float hn = go * tanhf(cn);
          if (s == 0) {
            int m = rbase + r;
            int h = (nts[i] << 2) + (nlo >> 2);
            h2buf[m * KPADH + h] = (__bf16)hn;
            if (t == T_STEPS - 1) h2out[(size_t)(m0 + m) * HDIM + h] = hn;
          }
        }
      }
    }
    // next iteration's first barrier provides the h2-write fence
  }
}

// --------------------------------------------------------------------------
// Kernel 3: head — softmax(relu(h2@Wd1+bd1)@Wd2+bd2), one block per batch row
__global__ void __launch_bounds__(256) k_head(const float* __restrict__ h2f,
                                              const float* __restrict__ Wd1,
                                              const float* __restrict__ bd1,
                                              const float* __restrict__ Wd2,
                                              const float* __restrict__ bd2,
                                              float* __restrict__ out) {
  __shared__ float sh_h[HDIM];
  __shared__ float sh_y[HDIM];
  __shared__ float sh_l[VDIM];
  __shared__ float red[256];
  const int tid = threadIdx.x;
  const int b = blockIdx.x;
  if (tid < HDIM) sh_h[tid] = h2f[(size_t)b * HDIM + tid];
  __syncthreads();
  if (tid < HDIM) {
    float s = bd1[tid];
    for (int k = 0; k < HDIM; ++k) s += sh_h[k] * Wd1[k * HDIM + tid];
    sh_y[tid] = fmaxf(s, 0.0f);
  }
  __syncthreads();
  for (int v = tid; v < VDIM; v += 256) {
    float s = bd2[v];
    for (int k = 0; k < HDIM; ++k) s += sh_y[k] * Wd2[k * VDIM + v];
    sh_l[v] = s;
  }
  __syncthreads();
  float m = -3.4e38f;
  for (int v = tid; v < VDIM; v += 256) m = fmaxf(m, sh_l[v]);
  red[tid] = m;
  __syncthreads();
  for (int off = 128; off > 0; off >>= 1) {
    if (tid < off) red[tid] = fmaxf(red[tid], red[tid + off]);
    __syncthreads();
  }
  float mx = red[0];
  __syncthreads();
  float ssum = 0.0f;
  for (int v = tid; v < VDIM; v += 256) {
    float e = __expf(sh_l[v] - mx);
    sh_l[v] = e;
    ssum += e;
  }
  red[tid] = ssum;
  __syncthreads();
  for (int off = 128; off > 0; off >>= 1) {
    if (tid < off) red[tid] += red[tid + off];
    __syncthreads();
  }
  float inv = 1.0f / red[0];
  for (int v = tid; v < VDIM; v += 256) out[(size_t)b * VDIM + v] = sh_l[v] * inv;
}

// --------------------------------------------------------------------------
extern "C" void kernel_launch(void* const* d_in, const int* in_sizes, int n_in,
                              void* d_out, int out_size, void* d_ws, size_t ws_size,
                              hipStream_t stream) {
  const float* X   = (const float*)d_in[0];
  const float* W1  = (const float*)d_in[1];
  const float* U1  = (const float*)d_in[2];
  const float* b1  = (const float*)d_in[3];
  const float* W2  = (const float*)d_in[4];
  const float* U2  = (const float*)d_in[5];
  const float* b2  = (const float*)d_in[6];
  const float* Wd1 = (const float*)d_in[7];
  const float* bd1 = (const float*)d_in[8];
  const float* Wd2 = (const float*)d_in[9];
  const float* bd2 = (const float*)d_in[10];
  float* out = (float*)d_out;

  // Workspace layout:
  //   [0,            512K) : W1 bf16, permuted [400][608]           (486,400 B)
  //   [512K,         1M  ) : final h2 f32 [B][H]                    (204,800 B)
  //   [1M,   1M + 419.4MB) : XW f32 [T*B][400]
  char* ws = (char*)d_ws;
  __bf16* W1bf = (__bf16*)ws;
  float*  h2f  = (float*)(ws + (512 << 10));
  float*  XW   = (float*)(ws + (1 << 20));

  const int RECLDS = (3 * H4 * KPADH + 2 * 16 * KPADH) * 2;  // 315,392 B < 320 KB WGP LDS
  (void)hipFuncSetAttribute(reinterpret_cast<const void*>(&k_lstm),
                            hipFuncAttributeMaxDynamicSharedMemorySize, RECLDS);

  k_prep_w1<<<(H4 * KPADX + 255) / 256, 256, 0, stream>>>(W1, W1bf);
  k_gemm_xw<<<(T_STEPS * BATCH) / 64, 256, 0, stream>>>(X, W1bf, b1, XW);
  k_lstm<<<BATCH / 16, 256, RECLDS, stream>>>(XW, U1, W2, U2, b2, h2f);
  k_head<<<BATCH, 256, 0, stream>>>(h2f, Wd1, bd1, Wd2, bd2, out);
}
